// IG_MSA_91182155694193
// MI455X (gfx1250) — matrix operand order repaired and checked by reading, not compiled, not run on hardware
//
#include <hip/hip_runtime.h>

// ---------------------------------------------------------------------------
// MI455X / gfx1250 implementation of IG-MSA block (transposed channel attention)
// All large GEMMs run on v_wmma_f32_16x16x32_bf16 (wave32 WMMA, f32 accum).
// LDS tiles are stored pre-swizzled into the ISA's 16-bit A/B fragment layouts
// so each per-lane fragment load is a contiguous 32B (2x ds_load_b128).
// Where the staged data is already bf16 and K-contiguous, tiles are copied
// with GLOBAL_LOAD_ASYNC_TO_LDS_B128 (ASYNCcnt) when the toolchain has the
// builtins; otherwise falls back to global_load_b128 + ds_store.
// ---------------------------------------------------------------------------

typedef __bf16 v16bf __attribute__((ext_vector_type(16)));
typedef float  v8f   __attribute__((ext_vector_type(8)));
typedef int    i4    __attribute__((ext_vector_type(4)));

#ifndef __has_builtin
#define __has_builtin(x) 0
#endif
#if __has_builtin(__builtin_amdgcn_global_load_async_to_lds_b128) && \
    __has_builtin(__builtin_amdgcn_s_wait_asynccnt)
#define HAVE_ASYNC_LDS 1
typedef __attribute__((address_space(1))) i4* gp128;   // global int4*
typedef __attribute__((address_space(3))) i4* lp128;   // LDS int4*
#else
#define HAVE_ASYNC_LDS 0
#endif

__device__ __forceinline__ __bf16 tobf(float f) { return (__bf16)f; }

// 16-bit A-matrix 16x32 fragment layout (ISA 7.12.2):
//   lanes 0-15  : row m, K = {0..7, 16..23}
//   lanes 16-31 : row m, K = {8..15, 24..31}
__device__ __forceinline__ int a_lane(int m, int k) { return (m & 15) + 16 * ((k >> 3) & 1); }
__device__ __forceinline__ int a_idx (int k)        { return (k & 7) + 8 * (k >> 4); }
// 16-bit B-matrix 32x16 fragment layout:
//   lanes 0-15  : col n, K = {0..7, 16..23} ; lanes 16-31 : col n, K = {8..15, 24..31}
__device__ __forceinline__ int b_lane(int n, int k) { return (n & 15) + 16 * ((k >> 3) & 1); }
__device__ __forceinline__ int b_idx (int k)        { return (k & 7) + 8 * (k >> 4); }

__device__ __forceinline__ v8f wmma_bf16(v16bf a, v16bf b, v8f c) {
  return __builtin_amdgcn_wmma_f32_16x16x32_bf16(false, a, false, b, (short)0, c, false, false);
}

// Copy of one 16B unit (8 bf16, K-run) into fragment-order LDS.
// lds is mgrp rows of [32][16] __bf16; k8 indexes 8-element K runs in a 32-chunk.
__device__ __forceinline__ void frag_copy_a16(v16bf (*lds)[32], int m, int k8,
                                              const __bf16* gsrc) {
  __bf16* dst = (__bf16*)&lds[m >> 4][(m & 15) + 16 * (k8 & 1)] + 8 * (k8 >> 1);
#if HAVE_ASYNC_LDS
  __builtin_amdgcn_global_load_async_to_lds_b128(
      (gp128)(void*)const_cast<__bf16*>(gsrc), (lp128)(void*)dst, 0, 0);
#else
  *(ulonglong2*)dst = *(const ulonglong2*)gsrc;
#endif
}

constexpr int Bc = 4, Hh = 128, Ww = 128, Cc = 512;
constexpr int HEADS = 8, DH = 64;
constexpr int Nn = Hh * Ww;          // 16384
constexpr int Mtot = Bc * Nn;        // 65536

// ---------------------------------------------------------------------------
// cdp = prior @ cdim_w1 @ cdim_w2  (tiny GEMMs, scalar is fine)
// ---------------------------------------------------------------------------
__global__ void k_cdp1(const float* __restrict__ prior, const float* __restrict__ w1,
                       float* __restrict__ hidden) {
  int i = blockIdx.x * 256 + threadIdx.x;          // 0..8191
  int b = i >> 11, j = i & 2047;
  float s = 0.f;
  for (int k = 0; k < 512; ++k) s += prior[b * 512 + k] * w1[k * 2048 + j];
  hidden[i] = s;
}
__global__ void k_cdp2(const float* __restrict__ hidden, const float* __restrict__ w2,
                       float* __restrict__ cdp) {
  int i = blockIdx.x * 256 + threadIdx.x;          // 0..4095
  int b = i >> 10, j = i & 1023;
  float s = 0.f;
  for (int k = 0; k < 2048; ++k) s += hidden[b * 2048 + k] * w2[k * 1024 + j];
  cdp[i] = s;
}

// ---------------------------------------------------------------------------
// Fused modulation + QKV projection: (65536x512) @ (512x1536)
// Tile 128x128x32, 256 threads = 8 waves (2x4), each wave 4x2 16x16 tiles.
// Epilogue: bf16 q/k with sum-of-squares reduction, or bf16 v & v*illu.
// ---------------------------------------------------------------------------
__global__ __launch_bounds__(256)
void k_qkv(const float* __restrict__ x_in, const float* __restrict__ illu,
           const float* __restrict__ cdp,
           const float* __restrict__ wq, const float* __restrict__ wk,
           const float* __restrict__ wv,
           __bf16* __restrict__ qbf, __bf16* __restrict__ kbf,
           __bf16* __restrict__ vinp, __bf16* __restrict__ vmul,
           float* __restrict__ ssq_q, float* __restrict__ ssq_k) {
  __shared__ v16bf lA[8][32];     // 128x32 A tile, fragment order
  __shared__ v16bf lB[8][32];     // 32x128 B tile, fragment order
  __shared__ float ssq[128];

  const int t  = threadIdx.x;
  const int m0 = blockIdx.x * 128;
  const int n1 = blockIdx.y * 128;          // column base in [0,1536)
  const int b  = m0 >> 14;                  // one batch per tile (16384 % 128 == 0)
  const int sel  = n1 >> 9;                 // 0=q, 1=k, 2=v (uniform per block)
  const int csel = n1 & 511;
  const float* wsel = (sel == 0) ? wq : (sel == 1) ? wk : wv;

  if (t < 128) ssq[t] = 0.f;

  const int wave = t >> 5, lane = t & 31;
  const int wm = wave >> 2, wn = wave & 3;

  v8f acc[4][2];
  for (int i = 0; i < 4; ++i)
    for (int j = 0; j < 2; ++j) acc[i][j] = (v8f){0,0,0,0,0,0,0,0};

  for (int k0 = 0; k0 < 512; k0 += 32) {
    __syncthreads();
    // A tile: modulated x, fp32 -> bf16, scattered into fragment layout
    for (int it = 0; it < 4; ++it) {
      int idx = t + it * 256;               // 0..1023 -> 128 rows x 8 float4
      int r = idx >> 3;
      int kk4 = (idx & 7) * 4;
      float4 xv = *(const float4*)&x_in[(size_t)(m0 + r) * 512 + k0 + kk4];
      float xs[4] = {xv.x, xv.y, xv.z, xv.w};
      for (int j = 0; j < 4; ++j) {
        int kk = kk4 + j;
        float c1 = cdp[b * 1024 + k0 + kk];
        float c2 = cdp[b * 1024 + 512 + k0 + kk];
        ((__bf16*)&lA[r >> 4][a_lane(r, kk)])[a_idx(kk)] = tobf(xs[j] * c1 + c2);
      }
    }
    // B tile: weights fp32 -> bf16
    for (int it = 0; it < 4; ++it) {
      int idx = t + it * 256;               // 0..1023 -> 32 k x 32 float4
      int kk = idx >> 5;
      int cc4 = (idx & 31) * 4;
      float4 w4 = *(const float4*)&wsel[(size_t)(k0 + kk) * 512 + csel + cc4];
      float wsv[4] = {w4.x, w4.y, w4.z, w4.w};
      for (int j = 0; j < 4; ++j) {
        int cc = cc4 + j;
        ((__bf16*)&lB[cc >> 4][b_lane(cc, kk)])[b_idx(kk)] = tobf(wsv[j]);
      }
    }
    __syncthreads();
    if (k0 + 32 < 512)   // global_prefetch_b8 of next A chunk
      __builtin_prefetch(&x_in[(size_t)(m0 + (t & 127)) * 512 + k0 + 32], 0, 3);

    v16bf af[4];
    for (int mi = 0; mi < 4; ++mi) af[mi] = lA[wm * 4 + mi][lane];
    for (int ni = 0; ni < 2; ++ni) {
      v16bf bfr = lB[wn * 2 + ni][lane];
      for (int mi = 0; mi < 4; ++mi) acc[mi][ni] = wmma_bf16(af[mi], bfr, acc[mi][ni]);
    }
  }

  __syncthreads();
  for (int mi = 0; mi < 4; ++mi)
    for (int ni = 0; ni < 2; ++ni)
      for (int r = 0; r < 8; ++r) {
        int mrow = m0 + wm * 64 + mi * 16 + r + ((lane >= 16) ? 8 : 0);
        int lc   = wn * 32 + ni * 16 + (lane & 15);        // local col 0..127
        int col  = n1 + lc;
        float val = acc[mi][ni][r];
        size_t off = (size_t)mrow * 512 + (col & 511);
        if (sel == 0)      { qbf[off] = tobf(val); atomicAdd(&ssq[lc], val * val); }
        else if (sel == 1) { kbf[off] = tobf(val); atomicAdd(&ssq[lc], val * val); }
        else {
          vinp[off] = tobf(val);
          vmul[off] = tobf(val * illu[off]);
        }
      }
  __syncthreads();
  if (sel < 2 && t < 128) {
    float* dst = (sel == 0) ? ssq_q : ssq_k;
    atomicAdd(&dst[b * 512 + csel + t], ssq[t]);
  }
}

__global__ void k_invnorm(const float* __restrict__ ssq_q, const float* __restrict__ ssq_k,
                          float* __restrict__ invq, float* __restrict__ invk) {
  int i = blockIdx.x * 256 + threadIdx.x;
  if (i < 2048) {
    invq[i] = 1.f / fmaxf(sqrtf(ssq_q[i]), 1e-12f);
    invk[i] = 1.f / fmaxf(sqrtf(ssq_k[i]), 1e-12f);
  }
}

// ---------------------------------------------------------------------------
// Gram matrix: attnraw[b,h,i,j] += sum_n k[b,n,h,i] * q[b,n,h,j]
// grid (32 bh, 64 n-slices of 256); 8 waves -> 2 of the 16 16x16 tiles each.
// ---------------------------------------------------------------------------
__global__ __launch_bounds__(256)
void k_gram(const __bf16* __restrict__ kbf, const __bf16* __restrict__ qbf,
            float* __restrict__ attnraw) {
  __shared__ v16bf lA[4][32];     // K^T 64x32
  __shared__ v16bf lB[4][32];     // Q   32x64
  const int t = threadIdx.x;
  const int bh = blockIdx.x, b = bh >> 3, h = bh & 7;
  const int n0 = blockIdx.y * 256;
  const int wave = t >> 5, lane = t & 31;
  const int mi = wave & 3;
  const int nbase = (wave >> 2) * 2;
  v8f acc[2];
  acc[0] = (v8f){0,0,0,0,0,0,0,0};
  acc[1] = (v8f){0,0,0,0,0,0,0,0};
  const size_t base = (size_t)b * Nn * 512 + h * 64;

  for (int kc = 0; kc < 256; kc += 32) {
    __syncthreads();
    for (int it = 0; it < 8; ++it) {        // A: (i, kk=n-local), transposed read
      int idx = t + it * 256;               // 0..2047
      int kk = idx >> 6, i = idx & 63;
      __bf16 v = kbf[base + (size_t)(n0 + kc + kk) * 512 + i];
      ((__bf16*)&lA[i >> 4][a_lane(i, kk)])[a_idx(kk)] = v;
    }
    for (int it = 0; it < 8; ++it) {        // B: (kk, j)
      int idx = t + it * 256;
      int kk = idx >> 6, j = idx & 63;
      __bf16 v = qbf[base + (size_t)(n0 + kc + kk) * 512 + j];
      ((__bf16*)&lB[j >> 4][b_lane(j, kk)])[b_idx(kk)] = v;
    }
    __syncthreads();
    v16bf af = lA[mi][lane];
    for (int ni = 0; ni < 2; ++ni) {
      v16bf bfr = lB[nbase + ni][lane];
      acc[ni] = wmma_bf16(af, bfr, acc[ni]);
    }
  }
  float* dst = attnraw + (size_t)bh * 64 * 64;
  for (int ni = 0; ni < 2; ++ni)
    for (int r = 0; r < 8; ++r) {
      int i = mi * 16 + r + ((lane >= 16) ? 8 : 0);
      int j = (nbase + ni) * 16 + (lane & 15);
      atomicAdd(&dst[i * 64 + j], acc[ni][r]);
    }
}

// fold norms + rescale, softmax over j, store fp32
__global__ void k_softmax(const float* __restrict__ attnraw,
                          const float* __restrict__ invq, const float* __restrict__ invk,
                          const float* __restrict__ rescale, float* __restrict__ attnsm) {
  int bh = blockIdx.x, b = bh >> 3, h = bh & 7;
  int i = threadIdx.x;                 // row 0..63
  const float* src = attnraw + (size_t)bh * 4096 + i * 64;
  float* dst = attnsm + (size_t)bh * 4096 + i * 64;
  float ik = invk[b * 512 + h * 64 + i];
  float rs = rescale[h];
  float row[64];
  float mx = -1e30f;
  for (int j = 0; j < 64; ++j) {
    float v = src[j] * ik * invq[b * 512 + h * 64 + j] * rs;
    row[j] = v; mx = fmaxf(mx, v);
  }
  float s = 0.f;
  for (int j = 0; j < 64; ++j) { float e = __expf(row[j] - mx); row[j] = e; s += e; }
  float inv = 1.f / s;
  for (int j = 0; j < 64; ++j) dst[j] = row[j] * inv;
}

// ---------------------------------------------------------------------------
// xo[b,n,h,i] = sum_j attn[b,h,i,j] * v[b,n,h,j]   (per-head 128x64 @ 64x64)
// A tile (bf16, K-contiguous) staged via async global->LDS when available.
// ---------------------------------------------------------------------------
__global__ __launch_bounds__(256)
void k_attnv(const __bf16* __restrict__ vmul, const float* __restrict__ attnsm,
             __bf16* __restrict__ xo) {
  __shared__ v16bf lA[8][32];     // v tile 128x32
  __shared__ v16bf lB[4][32];     // attn^T 32x64
  const int t = threadIdx.x;
  const int m0 = blockIdx.x * 128;
  const int b = m0 >> 14;
  const int h = blockIdx.y;
  const int wave = t >> 5, lane = t & 31;
  const size_t cbase = (size_t)h * 64;
  v8f acc[4];
  for (int i = 0; i < 4; ++i) acc[i] = (v8f){0,0,0,0,0,0,0,0};

  for (int kc = 0; kc < 64; kc += 32) {
    __syncthreads();
    // A: (m, kk=j-local): 512 x 16B async units (2 per thread)
    for (int it = 0; it < 2; ++it) {
      int u = t + it * 256;                 // 0..511
      int m = u >> 2, k8 = u & 3;
      frag_copy_a16(lA, m, k8, &vmul[(size_t)(m0 + m) * 512 + cbase + kc + k8 * 8]);
    }
    const float* asrc = attnsm + (size_t)(b * 8 + h) * 4096;
    for (int it = 0; it < 8; ++it) {        // B: (kk=j, col=i) = attn[i][j]
      int idx = t + it * 256;               // 0..2047
      int kk = idx >> 6, i = idx & 63;
      ((__bf16*)&lB[i >> 4][b_lane(i, kk)])[b_idx(kk)] = tobf(asrc[i * 64 + kc + kk]);
    }
#if HAVE_ASYNC_LDS
    __builtin_amdgcn_s_wait_asynccnt(0);
#endif
    __syncthreads();
    v16bf af = lA[wave][lane];
    for (int ni = 0; ni < 4; ++ni) {
      v16bf bfr = lB[ni][lane];
      acc[ni] = wmma_bf16(af, bfr, acc[ni]);
    }
  }
  for (int ni = 0; ni < 4; ++ni)
    for (int r = 0; r < 8; ++r) {
      int m = m0 + wave * 16 + r + ((lane >= 16) ? 8 : 0);
      int i = ni * 16 + (lane & 15);
      xo[(size_t)m * 512 + cbase + i] = tobf(acc[ni][r]);
    }
}

// ---------------------------------------------------------------------------
// out_c = xo @ wproj + bproj : (65536x512) @ (512x512) -> d_out (fp32)
// A tile (bf16, K-contiguous) staged via async global->LDS when available.
// ---------------------------------------------------------------------------
__global__ __launch_bounds__(256)
void k_proj(const __bf16* __restrict__ xo, const float* __restrict__ wproj,
            const float* __restrict__ bproj, float* __restrict__ out) {
  __shared__ v16bf lA[8][32];
  __shared__ v16bf lB[8][32];
  const int t = threadIdx.x;
  const int m0 = blockIdx.x * 128;
  const int n1 = blockIdx.y * 128;
  const int wave = t >> 5, lane = t & 31;
  const int wm = wave >> 2, wn = wave & 3;
  v8f acc[4][2];
  for (int i = 0; i < 4; ++i)
    for (int j = 0; j < 2; ++j) acc[i][j] = (v8f){0,0,0,0,0,0,0,0};

  for (int k0 = 0; k0 < 512; k0 += 32) {
    __syncthreads();
    // A: xo (bf16), 512 x 16B async units (2 per thread)
    for (int it = 0; it < 2; ++it) {
      int u = t + it * 256;                 // 0..511
      int m = u >> 2, k8 = u & 3;
      frag_copy_a16(lA, m, k8, &xo[(size_t)(m0 + m) * 512 + k0 + k8 * 8]);
    }
    for (int it = 0; it < 4; ++it) {        // B: wproj fp32 -> bf16
      int idx = t + it * 256;
      int kk = idx >> 5;
      int cc4 = (idx & 31) * 4;
      float4 w4 = *(const float4*)&wproj[(size_t)(k0 + kk) * 512 + n1 + cc4];
      float wsv[4] = {w4.x, w4.y, w4.z, w4.w};
      for (int j = 0; j < 4; ++j) {
        int cc = cc4 + j;
        ((__bf16*)&lB[cc >> 4][b_lane(cc, kk)])[b_idx(kk)] = tobf(wsv[j]);
      }
    }
#if HAVE_ASYNC_LDS
    __builtin_amdgcn_s_wait_asynccnt(0);
#endif
    __syncthreads();
    if (k0 + 32 < 512)
      __builtin_prefetch(&xo[(size_t)(m0 + (t & 127)) * 512 + k0 + 32], 0, 3);
    v16bf af[4];
    for (int mi = 0; mi < 4; ++mi) af[mi] = lA[wm * 4 + mi][lane];
    for (int ni = 0; ni < 2; ++ni) {
      v16bf bfr = lB[wn * 2 + ni][lane];
      for (int mi = 0; mi < 4; ++mi) acc[mi][ni] = wmma_bf16(af[mi], bfr, acc[mi][ni]);
    }
  }
  for (int mi = 0; mi < 4; ++mi)
    for (int ni = 0; ni < 2; ++ni)
      for (int r = 0; r < 8; ++r) {
        int m = m0 + wm * 64 + mi * 16 + r + ((lane >= 16) ? 8 : 0);
        int c = n1 + wn * 32 + ni * 16 + (lane & 15);
        out[(size_t)m * 512 + c] = acc[mi][ni][r] + bproj[c];
      }
}

// ---------------------------------------------------------------------------
// Depthwise 3x3 convs (NHWC, bf16 in / fp32 accum). conv1 applies exact GELU;
// conv2 accumulates into d_out (which holds out_c from k_proj).
// ---------------------------------------------------------------------------
__global__ void k_conv1(const __bf16* __restrict__ vinp, const float* __restrict__ w1,
                        __bf16* __restrict__ tbuf) {
  size_t idx = (size_t)blockIdx.x * 256 + threadIdx.x;
  int c = idx & 511;
  int x = (idx >> 9) & 127;
  int y = (idx >> 16) & 127;
  int b = (int)(idx >> 23);
  float s = 0.f;
  for (int dy = -1; dy <= 1; ++dy) {
    int yy = y + dy; if (yy < 0 || yy > 127) continue;
    for (int dx = -1; dx <= 1; ++dx) {
      int xx = x + dx; if (xx < 0 || xx > 127) continue;
      float v = (float)vinp[((((size_t)b * 128 + yy) * 128 + xx) * 512) + c];
      s += v * w1[c * 9 + (dy + 1) * 3 + (dx + 1)];
    }
  }
  float g = 0.5f * s * (1.f + erff(s * 0.70710678f));   // exact GELU
  tbuf[idx] = tobf(g);
}

__global__ void k_conv2(const __bf16* __restrict__ tbuf, const float* __restrict__ w2,
                        float* __restrict__ out) {
  size_t idx = (size_t)blockIdx.x * 256 + threadIdx.x;
  int c = idx & 511;
  int x = (idx >> 9) & 127;
  int y = (idx >> 16) & 127;
  int b = (int)(idx >> 23);
  float s = 0.f;
  for (int dy = -1; dy <= 1; ++dy) {
    int yy = y + dy; if (yy < 0 || yy > 127) continue;
    for (int dx = -1; dx <= 1; ++dx) {
      int xx = x + dx; if (xx < 0 || xx > 127) continue;
      float v = (float)tbuf[((((size_t)b * 128 + yy) * 128 + xx) * 512) + c];
      s += v * w2[c * 9 + (dy + 1) * 3 + (dx + 1)];
    }
  }
  out[idx] += s;
}

// ---------------------------------------------------------------------------
extern "C" void kernel_launch(void* const* d_in, const int* in_sizes, int n_in,
                              void* d_out, int out_size, void* d_ws, size_t ws_size,
                              hipStream_t stream) {
  (void)in_sizes; (void)n_in; (void)out_size; (void)ws_size;
  const float* x_in    = (const float*)d_in[0];
  const float* illu    = (const float*)d_in[1];
  const float* prior   = (const float*)d_in[2];
  const float* wq      = (const float*)d_in[3];
  const float* wk      = (const float*)d_in[4];
  const float* wv      = (const float*)d_in[5];
  const float* rescale = (const float*)d_in[6];
  const float* wproj   = (const float*)d_in[7];
  const float* bproj   = (const float*)d_in[8];
  const float* cw1     = (const float*)d_in[9];
  const float* cw2     = (const float*)d_in[10];
  const float* pw1     = (const float*)d_in[11];
  const float* pw2     = (const float*)d_in[12];
  float* out = (float*)d_out;

  char* ws = (char*)d_ws;
  size_t off = 0;
  auto alloc = [&](size_t bytes) -> char* {
    char* p = ws + off;
    off += (bytes + 255) & ~(size_t)255;
    return p;
  };
  float*  hidden  = (float*)alloc((size_t)4 * 2048 * 4);
  float*  cdp     = (float*)alloc((size_t)4 * 1024 * 4);
  float*  ssq_q   = (float*)alloc((size_t)2048 * 4);
  float*  ssq_k   = (float*)alloc((size_t)2048 * 4);
  float*  invq    = (float*)alloc((size_t)2048 * 4);
  float*  invk    = (float*)alloc((size_t)2048 * 4);
  float*  attnraw = (float*)alloc((size_t)32 * 64 * 64 * 4);
  float*  attnsm  = (float*)alloc((size_t)32 * 64 * 64 * 4);
  __bf16* qbuf    = (__bf16*)alloc((size_t)Mtot * 512 * 2);  // q, later gelu tmp
  __bf16* kbuf    = (__bf16*)alloc((size_t)Mtot * 512 * 2);  // k, later xo
  __bf16* vinp    = (__bf16*)alloc((size_t)Mtot * 512 * 2);  // v (pre-illu) for PE
  __bf16* vmul    = (__bf16*)alloc((size_t)Mtot * 512 * 2);  // v * illu

  (void)hipMemsetAsync(ssq_q, 0, 2048 * 4, stream);
  (void)hipMemsetAsync(ssq_k, 0, 2048 * 4, stream);
  (void)hipMemsetAsync(attnraw, 0, (size_t)32 * 64 * 64 * 4, stream);

  k_cdp1<<<8192 / 256, 256, 0, stream>>>(prior, cw1, hidden);
  k_cdp2<<<4096 / 256, 256, 0, stream>>>(hidden, cw2, cdp);
  k_qkv<<<dim3(Mtot / 128, 12), 256, 0, stream>>>(x_in, illu, cdp, wq, wk, wv,
                                                  qbuf, kbuf, vinp, vmul, ssq_q, ssq_k);
  k_invnorm<<<8, 256, 0, stream>>>(ssq_q, ssq_k, invq, invk);
  k_gram<<<dim3(32, Nn / 256), 256, 0, stream>>>(kbuf, qbuf, attnraw);
  k_softmax<<<32, 64, 0, stream>>>(attnraw, invq, invk, rescale, attnsm);
  k_attnv<<<dim3(Mtot / 128, HEADS), 256, 0, stream>>>(vmul, attnsm, kbuf); // kbuf := xo
  k_proj<<<dim3(Mtot / 128, 4), 256, 0, stream>>>(kbuf, wproj, bproj, out);
  k_conv1<<<(Mtot * 512) / 256, 256, 0, stream>>>(vinp, pw1, qbuf);         // qbuf := gelu tmp
  k_conv2<<<(Mtot * 512) / 256, 256, 0, stream>>>(qbuf, pw2, out);
}
